// RobustGLA_56186762166687
// MI455X (gfx1250) — compile-verified, matrix operand order checked by reading, not run on hardware
//
#include <hip/hip_runtime.h>
#include <hip/hip_bf16.h>

#define BB 4
#define LL 2048
#define DD 1024
#define HH 8
#define DHD 128
#define NCH (LL / 64)

typedef __attribute__((ext_vector_type(16))) __bf16 v16bf;
typedef __attribute__((ext_vector_type(8))) float v8f;

union FragBF { v16bf v; uint4 q[2]; };

__device__ __forceinline__ v8f zero8() {
  v8f r;
#pragma unroll
  for (int i = 0; i < 8; i++) r[i] = 0.f;
  return r;
}

// Low 32 bits of a flat pointer to __shared__ == LDS byte offset (ISA 10.2).
__device__ __forceinline__ unsigned lds_off(const void* p) {
  return (unsigned)(uintptr_t)p;
}

// 16B/lane async copy global -> LDS (ASYNCcnt-tracked, no VGPR round trip).
__device__ __forceinline__ void async_cp16(unsigned ldsOff, const void* g) {
  asm volatile("global_load_async_to_lds_b128 %0, %1, off"
               :: "v"(ldsOff), "v"(g) : "memory");
}

// Load a 16x32 bf16 WMMA A/B fragment from row-major LDS [rows][ld].
// Layout (ISA 7.12.2): lanes 0-15 row=lane, K halves {0..7,16..23};
// lanes 16-31 row=lane-16, K halves {8..15,24..31}.
__device__ __forceinline__ v16bf frag_ld(const __bf16* base, int ld, int r0, int k0) {
  int lane = threadIdx.x & 31;
  int r = r0 + (lane & 15);
  int kb = k0 + ((lane & 16) ? 8 : 0);
  const __bf16* p = base + r * ld + kb;
  FragBF f;
  f.q[0] = *(const uint4*)(p);
  f.q[1] = *(const uint4*)(p + 16);
  return f.v;
}

// Same fragment but sourced from fp32 LDS (state matrix), converted to bf16.
__device__ __forceinline__ v16bf frag_ld_f32(const float* base, int ld, int r0, int k0) {
  int lane = threadIdx.x & 31;
  int r = r0 + (lane & 15);
  int kb = k0 + ((lane & 16) ? 8 : 0);
  const float* p = base + r * ld + kb;
  v16bf out;
#pragma unroll
  for (int e = 0; e < 8; e++) {
    out[e]     = (__bf16)p[e];
    out[e + 8] = (__bf16)p[16 + e];
  }
  return out;
}

__device__ __forceinline__ v8f wmma_bf16(v16bf a, v16bf b, v8f c) {
  return __builtin_amdgcn_wmma_f32_16x16x32_bf16(false, a, false, b, (short)0, c, false, false);
}

// Double-buffered async-staged 128x128xK GEMM core (BK=32, 8 waves, 2x4 tiles/wave).
// gA, gB: row-major bf16 with row stride 1024. Result in acc[2][4].
__device__ __forceinline__ void gemm_core(const __bf16* __restrict__ gA,
                                          const __bf16* __restrict__ gB,
                                          __bf16* As /*[2][4096]*/,
                                          __bf16* Bs /*[2][4096]*/,
                                          v8f (&acc)[2][4]) {
  int tid = threadIdx.x;
  int wid = tid >> 5;
  int wm = (wid >> 1) * 32, wn = (wid & 1) * 64;
  unsigned asBase = lds_off(As);
  unsigned bsBase = lds_off(Bs);
#pragma unroll
  for (int i = 0; i < 2; i++)
#pragma unroll
    for (int j = 0; j < 4; j++) acc[i][j] = zero8();

  // prologue: stage k-slice 0 into buffer 0
#pragma unroll
  for (int i = 0; i < 2; i++) {
    int ci = tid + i * 256;               // 512 chunks of 16B per tile
    int row = ci >> 2, c8 = (ci & 3) * 8;
    async_cp16(asBase + ci * 16u, gA + row * 1024 + c8);
    async_cp16(bsBase + ci * 16u, gB + row * 1024 + c8);
  }

  for (int kt = 0; kt < 32; kt++) {
    int buf = kt & 1;
    if (kt + 1 < 32) {
      int nb = (kt + 1) & 1;
      int k0 = (kt + 1) * 32;
#pragma unroll
      for (int i = 0; i < 2; i++) {
        int ci = tid + i * 256;
        int row = ci >> 2, c8 = (ci & 3) * 8;
        async_cp16(asBase + nb * 8192u + ci * 16u, gA + row * 1024 + k0 + c8);
        async_cp16(bsBase + nb * 8192u + ci * 16u, gB + row * 1024 + k0 + c8);
      }
      asm volatile("s_wait_asynccnt 0x4" ::: "memory");  // slice kt landed
    } else {
      asm volatile("s_wait_asynccnt 0x0" ::: "memory");
    }
    __syncthreads();  // slice kt visible to all waves
    const __bf16* Ab = As + buf * 4096;
    const __bf16* Bb = Bs + buf * 4096;
    v16bf a0 = frag_ld(Ab, 32, wm, 0);
    v16bf a1 = frag_ld(Ab, 32, wm + 16, 0);
#pragma unroll
    for (int j = 0; j < 4; j++) {
      v16bf b = frag_ld(Bb, 32, wn + j * 16, 0);
      acc[0][j] = wmma_bf16(a0, b, acc[0][j]);
      acc[1][j] = wmma_bf16(a1, b, acc[1][j]);
    }
    __syncthreads();  // release buffer for slice kt+2
  }
}

// ---------------- Kernel W: weights fp32 -> bf16 (one shot) ----------------
__global__ __launch_bounds__(256) void wconv_kernel(const float* __restrict__ w0,
                                                    const float* __restrict__ w1,
                                                    const float* __restrict__ w2,
                                                    const float* __restrict__ w3,
                                                    const float* __restrict__ w4,
                                                    __bf16* __restrict__ dst) {
  int mi = blockIdx.y;
  const float* w = (mi == 0) ? w0 : (mi == 1) ? w1 : (mi == 2) ? w2 : (mi == 3) ? w3 : w4;
  size_t i0 = ((size_t)blockIdx.x * 256 + threadIdx.x) * 4;
  __bf16* d = dst + (size_t)mi * (DD * DD);
  float4 f = *(const float4*)(w + i0);
  d[i0 + 0] = (__bf16)f.x;
  d[i0 + 1] = (__bf16)f.y;
  d[i0 + 2] = (__bf16)f.z;
  d[i0 + 3] = (__bf16)f.w;
}

// ---------------- Kernel 0: LayerNorm, fp32 -> bf16 ----------------
__global__ __launch_bounds__(256) void ln_kernel(const float* __restrict__ x,
                                                 const float* __restrict__ gamma,
                                                 const float* __restrict__ beta,
                                                 __bf16* __restrict__ xnb) {
  int row = blockIdx.x;  // 0..8191
  const float* xr = x + (size_t)row * DD;
  int tid = threadIdx.x;
  float s = 0.f, s2 = 0.f;
  float vals[4];
#pragma unroll
  for (int i = 0; i < 4; i++) {
    float v = xr[tid + 256 * i];
    vals[i] = v;
    s += v;
    s2 += v * v;
  }
  __shared__ float red0[8], red1[8];
#pragma unroll
  for (int off = 16; off; off >>= 1) {
    s += __shfl_down(s, off, 32);
    s2 += __shfl_down(s2, off, 32);
  }
  int wid = tid >> 5;
  if ((tid & 31) == 0) { red0[wid] = s; red1[wid] = s2; }
  __syncthreads();
  if (tid == 0) {
    float a = 0.f, b = 0.f;
#pragma unroll
    for (int w = 0; w < 8; w++) { a += red0[w]; b += red1[w]; }
    red0[0] = a / DD;
    red1[0] = b / DD;
  }
  __syncthreads();
  float mu = red0[0];
  float var = red1[0] - mu * mu;
  float rstd = rsqrtf(var + 1e-5f);
#pragma unroll
  for (int i = 0; i < 4; i++) {
    int c = tid + 256 * i;
    float xn = (vals[i] - mu) * rstd * gamma[c] + beta[c];
    xnb[(size_t)row * DD + c] = (__bf16)xn;
  }
}

// ------- Kernel 1: fused q/k/v/g projection GEMM (async-staged) -------
__global__ __launch_bounds__(256) void qkvg_kernel(const __bf16* __restrict__ xnb,
                                                   const __bf16* __restrict__ Wall,
                                                   __bf16* __restrict__ qo,
                                                   __bf16* __restrict__ ko,
                                                   __bf16* __restrict__ vo,
                                                   __bf16* __restrict__ go) {
  __shared__ __align__(16) __bf16 As[2 * 128 * 32];
  __shared__ __align__(16) __bf16 Bs[2 * 128 * 32];
  int m0 = blockIdx.y * 128;
  int ng = blockIdx.x * 128;
  int mid = ng >> 10;
  int j0 = ng & 1023;
  int tid = threadIdx.x, lane = tid & 31, wid = tid >> 5;
  int wm = (wid >> 1) * 32, wn = (wid & 1) * 64;

  v8f acc[2][4];
  gemm_core(xnb + (size_t)m0 * DD,
            Wall + (size_t)mid * (DD * DD) + (size_t)j0 * DD,
            As, Bs, acc);

  __bf16* outp = (mid == 0) ? qo : (mid == 1) ? ko : (mid == 2) ? vo : go;
#pragma unroll
  for (int i = 0; i < 2; i++)
#pragma unroll
    for (int j = 0; j < 4; j++)
#pragma unroll
      for (int r = 0; r < 8; r++) {
        int m = m0 + wm + i * 16 + r + ((lane & 16) ? 8 : 0);
        int n = j0 + wn + j * 16 + (lane & 15);
        float v = acc[i][j][r];
        if (mid == 3) v = fminf(v, 0.f) - __logf(1.f + __expf(-fabsf(v)));  // logsigmoid
        int b = m >> 11, l = m & (LL - 1);
        int h = n >> 7, dh = n & 127;
        outp[(((size_t)b * HH + h) * LL + l) * DHD + dh] = (__bf16)v;
      }
}

// ---------------- Kernel 2: chunked GLA scan, one block per (b,h) ----------------
// Chunk C=64. Chunk-local cumulative gate G (inclusive):
//   q~ = q*exp(G)*scale ; k~ = k*exp(-G) ; S_end = exp(Gtot)*(S0 + k~^T v)
//   O = mask(q~ k~^T) v + q~ S0
__global__ __launch_bounds__(256) void gla_kernel(const __bf16* __restrict__ q,
                                                  const __bf16* __restrict__ k,
                                                  const __bf16* __restrict__ v,
                                                  const __bf16* __restrict__ g,
                                                  __bf16* __restrict__ yb,
                                                  float* __restrict__ fs) {
  __shared__ __align__(16) __bf16 qt[64 * 128];   // [t][k]
  __shared__ __align__(16) __bf16 kc[64 * 128];   // [s][k]
  __shared__ __align__(16) __bf16 kcT[128 * 64];  // [k][s]
  __shared__ __align__(16) __bf16 vT[128 * 64];   // [v][s]
  __shared__ __align__(16) __bf16 At[64 * 64];    // masked intra scores [t][s]
  __shared__ __align__(16) float S[128 * 128];    // state, stored [v][k]
  __shared__ float eG[128];                       // exp(Gtot) per k-dim

  int bh = blockIdx.x;  // 0..31
  int b = bh >> 3, h = bh & 7;
  size_t base = (size_t)bh * LL * DHD;
  int tid = threadIdx.x, lane = tid & 31, wid = tid >> 5;
  const float scale = 0.08838834764831845f;  // 128^-0.5

  for (int i = tid; i < 128 * 128; i += 256) S[i] = 0.f;
  __syncthreads();

  for (int ci = 0; ci < NCH; ci++) {
    int t0 = ci * 64;
    // phase 1: serial gate scan (threads 0..127) + vT staging (threads 128..255)
    if (tid < 128) {
      int kd = tid;
      float G = 0.f;
      for (int t = 0; t < 64; t++) {
        size_t off = base + (size_t)(t0 + t) * DHD + kd;
        G += (float)g[off];
        float qv = (float)q[off];
        float kv = (float)k[off];
        qt[t * 128 + kd] = (__bf16)(qv * __expf(G) * scale);
        __bf16 kk = (__bf16)(kv * __expf(-G));
        kc[t * 128 + kd] = kk;
        kcT[kd * 64 + t] = kk;
      }
      eG[kd] = __expf(G);
    } else {
      int vd = tid - 128;
      for (int t = 0; t < 64; t++)
        vT[vd * 64 + t] = v[base + (size_t)(t0 + t) * DHD + vd];
    }
    __syncthreads();

    // phase 2: At = q~ x k~^T (64x64, K=128), causal mask, store bf16
    {
      int tm = wid >> 1;
      int tn0 = (wid & 1) * 2;
#pragma unroll
      for (int j = 0; j < 2; j++) {
        v8f acc = zero8();
#pragma unroll
        for (int kk = 0; kk < 4; kk++) {
          v16bf a = frag_ld(qt, 128, tm * 16, kk * 32);
          v16bf bf = frag_ld(kc, 128, (tn0 + j) * 16, kk * 32);
          acc = wmma_bf16(a, bf, acc);
        }
#pragma unroll
        for (int r = 0; r < 8; r++) {
          int t = tm * 16 + r + ((lane & 16) ? 8 : 0);
          int s = (tn0 + j) * 16 + (lane & 15);
          At[t * 64 + s] = (__bf16)((s <= t) ? acc[r] : 0.f);
        }
      }
    }
    __syncthreads();

    // phase 3: O = At x v + q~ x S_old  (64x128), store to y (bf16)
    {
      int tm = wid >> 1;
      int tn0 = (wid & 1) * 4;
      v8f acc[4];
#pragma unroll
      for (int j = 0; j < 4; j++) acc[j] = zero8();
#pragma unroll
      for (int kk = 0; kk < 2; kk++) {
        v16bf a = frag_ld(At, 64, tm * 16, kk * 32);
#pragma unroll
        for (int j = 0; j < 4; j++) {
          v16bf bf = frag_ld(vT, 64, (tn0 + j) * 16, kk * 32);
          acc[j] = wmma_bf16(a, bf, acc[j]);
        }
      }
#pragma unroll
      for (int kk = 0; kk < 4; kk++) {
        v16bf a = frag_ld(qt, 128, tm * 16, kk * 32);
#pragma unroll
        for (int j = 0; j < 4; j++) {
          v16bf bf = frag_ld_f32(S, 128, (tn0 + j) * 16, kk * 32);
          acc[j] = wmma_bf16(a, bf, acc[j]);
        }
      }
#pragma unroll
      for (int j = 0; j < 4; j++)
#pragma unroll
        for (int r = 0; r < 8; r++) {
          int t = tm * 16 + r + ((lane & 16) ? 8 : 0);
          int vd = (tn0 + j) * 16 + (lane & 15);
          yb[((size_t)b * LL + (t0 + t)) * DD + h * DHD + vd] = (__bf16)acc[j][r];
        }
    }
    __syncthreads();

    // phase 4: S = exp(Gtot) * (S + k~^T x v); wave 'wid' owns k-rows [16w,16w+16)
    {
      int tk = wid;
      int hi = (lane & 16) ? 8 : 0;
      int kbase = tk * 16 + hi;
      float eg[8];
#pragma unroll
      for (int r = 0; r < 8; r++) eg[r] = eG[kbase + r];
#pragma unroll
      for (int tv = 0; tv < 8; tv++) {
        float* sp = &S[((tv * 16) + (lane & 15)) * 128 + kbase];
        v8f c;
#pragma unroll
        for (int r = 0; r < 8; r++) c[r] = sp[r];
#pragma unroll
        for (int kk = 0; kk < 2; kk++) {
          v16bf a = frag_ld(kcT, 64, tk * 16, kk * 32);
          v16bf bf = frag_ld(vT, 64, tv * 16, kk * 32);
          c = wmma_bf16(a, bf, c);
        }
#pragma unroll
        for (int r = 0; r < 8; r++) sp[r] = c[r] * eg[r];
      }
    }
    __syncthreads();
  }

  // final_state [b][h][k][v] (S is stored [v][k])
  for (int i = tid; i < 128 * 128; i += 256) {
    int kd = i >> 7, vd = i & 127;
    fs[(size_t)bh * 128 * 128 + i] = S[vd * 128 + kd];
  }
}

// ---------------- Kernel 3: output projection (async-staged), fp32 result ----------------
__global__ __launch_bounds__(256) void outproj_kernel(const __bf16* __restrict__ yb,
                                                      const __bf16* __restrict__ Wob,
                                                      float* __restrict__ out) {
  __shared__ __align__(16) __bf16 As[2 * 128 * 32];
  __shared__ __align__(16) __bf16 Bs[2 * 128 * 32];
  int m0 = blockIdx.y * 128;
  int n0 = blockIdx.x * 128;
  int tid = threadIdx.x, lane = tid & 31, wid = tid >> 5;
  int wm = (wid >> 1) * 32, wn = (wid & 1) * 64;

  v8f acc[2][4];
  gemm_core(yb + (size_t)m0 * DD, Wob + (size_t)n0 * DD, As, Bs, acc);

#pragma unroll
  for (int i = 0; i < 2; i++)
#pragma unroll
    for (int j = 0; j < 4; j++)
#pragma unroll
      for (int r = 0; r < 8; r++) {
        int m = m0 + wm + i * 16 + r + ((lane & 16) ? 8 : 0);
        int n = n0 + wn + j * 16 + (lane & 15);
        out[(size_t)m * DD + n] = acc[i][j][r];
      }
}

extern "C" void kernel_launch(void* const* d_in, const int* in_sizes, int n_in,
                              void* d_out, int out_size, void* d_ws, size_t ws_size,
                              hipStream_t stream) {
  const float* x = (const float*)d_in[0];
  const float* gamma = (const float*)d_in[1];
  const float* beta = (const float*)d_in[2];
  const float* Wq = (const float*)d_in[3];
  const float* Wk = (const float*)d_in[4];
  const float* Wv = (const float*)d_in[5];
  const float* Wg = (const float*)d_in[6];
  const float* Wo = (const float*)d_in[7];

  char* ws = (char*)d_ws;
  const size_t SZ = (size_t)BB * LL * DD * sizeof(__bf16);  // 16 MB
  __bf16* xnb = (__bf16*)ws;            // reused as y after the scan
  __bf16* qb = (__bf16*)(ws + SZ);
  __bf16* kb = (__bf16*)(ws + 2 * SZ);
  __bf16* vb = (__bf16*)(ws + 3 * SZ);
  __bf16* gb = (__bf16*)(ws + 4 * SZ);
  __bf16* wb = (__bf16*)(ws + 5 * SZ);  // 5 x 1024x1024 bf16 weights (q,k,v,g,o)

  float* out = (float*)d_out;
  float* fs = out + (size_t)BB * LL * DD;  // final_state region

  wconv_kernel<<<dim3(1024, 5), 256, 0, stream>>>(Wq, Wk, Wv, Wg, Wo, wb);
  ln_kernel<<<BB * LL, 256, 0, stream>>>(x, gamma, beta, xnb);
  qkvg_kernel<<<dim3(32, 64), 256, 0, stream>>>(xnb, wb, qb, kb, vb, gb);
  gla_kernel<<<BB * HH, 256, 0, stream>>>(qb, kb, vb, gb, xnb /*y*/, fs);
  outproj_kernel<<<dim3(8, 64), 256, 0, stream>>>(xnb /*y*/, wb + 4 * (size_t)(DD * DD), out);
}
